// Q_input_proj_12876311953736
// MI455X (gfx1250) — 8-bit matrix operand order repaired and checked by reading, compiled, not run on hardware
//
#include <hip/hip_runtime.h>


typedef __attribute__((ext_vector_type(8))) int v8i;

#define WS_AMAX 0
#define WS_SW   256
#define WS_A    4096
#define WS_X    (4096 + 256 * 2048)

__device__ __forceinline__ float scale_a_from_bits(unsigned bits) {
    return fmaxf(__uint_as_float(bits), 1e-8f) * (1.0f / 127.0f);
}

__global__ void init_kernel(unsigned* amaxBits) { *amaxBits = 0u; }

__global__ void amax_kernel(const float* __restrict__ x, unsigned* amaxBits, int n4) {
    const float4* x4 = (const float4*)x;
    int stride = gridDim.x * blockDim.x;
    float am = 0.0f;
    for (int i = blockIdx.x * blockDim.x + threadIdx.x; i < n4; i += stride) {
        float4 v = x4[i];
        am = fmaxf(am, fmaxf(fmaxf(fabsf(v.x), fabsf(v.y)),
                             fmaxf(fabsf(v.z), fabsf(v.w))));
    }
    for (int off = 16; off > 0; off >>= 1)
        am = fmaxf(am, __shfl_xor(am, off, 32));
    __shared__ float red[8];
    int lane = threadIdx.x & 31, wv = threadIdx.x >> 5;
    if (lane == 0) red[wv] = am;
    __syncthreads();
    if (wv == 0) {
        am = (lane < (int)(blockDim.x >> 5)) ? red[lane] : 0.0f;
        for (int off = 4; off > 0; off >>= 1)
            am = fmaxf(am, __shfl_xor(am, off, 32));
        if (lane == 0) atomicMax(amaxBits, __float_as_uint(am));
    }
}

__global__ void wquant_kernel(const float* __restrict__ w,
                              float* __restrict__ s_w_out,
                              char* __restrict__ Aq) {
    int co = blockIdx.x;
    int t  = threadIdx.x;
    const float* row = w + (size_t)co * 2048;
    float vals[8];
    float am = 0.0f;
#pragma unroll
    for (int i = 0; i < 8; ++i) {
        vals[i] = row[t + 256 * i];
        am = fmaxf(am, fabsf(vals[i]));
    }
    __shared__ float red[256];
    red[t] = am;
    __syncthreads();
    for (int s = 128; s > 0; s >>= 1) {
        if (t < s) red[t] = fmaxf(red[t], red[t + s]);
        __syncthreads();
    }
    float s_w = fmaxf(red[0], 1e-8f) * (1.0f / 127.0f);
    if (t == 0) s_w_out[co] = s_w;
    float inv = 1.0f / s_w;
    int m = co & 15, mt = co >> 4;
#pragma unroll
    for (int i = 0; i < 8; ++i) {
        int k = t + 256 * i;
        int q = (int)rintf(vals[i] * inv);
        q = max(-128, min(127, q));
        int kt = k >> 6, kk = k & 63;
        int j = kk & 3, u = kk >> 2;
        int v_ = ((u >> 3) << 2) | (u & 3);
        int hi = (u >> 2) & 1;
        size_t off = ((size_t)mt * 32 + kt) * 1024 + (size_t)(m + 16 * hi) * 32
                   + v_ * 4 + j;
        Aq[off] = (char)q;
    }
}

__global__ void xquant_kernel(const float* __restrict__ x,
                              const unsigned* __restrict__ amaxBits,
                              char* __restrict__ Xq) {
    int t = blockIdx.x * blockDim.x + threadIdx.x;
    int p   = t & 1023;
    int ci4 = (t >> 10) & 511;
    int b   = t >> 19;
    int ci  = ci4 * 4;
    float inv = 127.0f / fmaxf(__uint_as_float(*amaxBits), 1e-8f);
    const float* xp = x + ((size_t)b * 2048 + ci) * 1024 + p;
    unsigned packed = 0;
#pragma unroll
    for (int i = 0; i < 4; ++i) {
        int q = (int)rintf(xp[(size_t)i * 1024] * inv);
        q = max(-128, min(127, q));
        packed |= (unsigned)(q & 0xff) << (8 * i);
    }
    int n  = b * 1024 + p;
    int nt = n >> 4, nn = n & 15;
    int kt = ci >> 6, kk = ci & 63;
    int u  = kk >> 2;
    int v_ = ((u >> 3) << 2) | (u & 3);
    int hi = (u >> 2) & 1;
    size_t off = ((size_t)nt * 32 + kt) * 1024 + (size_t)(nn + 16 * hi) * 32 + v_ * 4;
    *(unsigned*)(Xq + off) = packed;
}

__global__ void __launch_bounds__(256)
gemm_iu8_kernel(const char* __restrict__ Aq,
                const char* __restrict__ Xq,
                const float* __restrict__ s_w,
                const float* __restrict__ bias,
                const unsigned* __restrict__ amaxBits,
                float* __restrict__ out) {
    __shared__ __align__(128) char smem[4096];

    int t    = threadIdx.x;
    int lane = t & 31;
    int wv   = t >> 5;
    int mtp  = wv;
    int ntp  = blockIdx.x;
    int mt0  = mtp * 2, nt0 = ntp * 2;

    const char* aBase0 = Aq + ((size_t)mt0 * 32) * 1024 + (size_t)lane * 32;
    const char* aBase1 = aBase0 + 32 * 1024;

    unsigned smemBase = (unsigned)(uintptr_t)(&smem[0]);
    const char* bStage = Xq + ((size_t)(nt0 + (t >> 7)) * 32) * 1024
                            + (size_t)(t & 127) * 8;
    auto stage = [&](int kt, int buf) {
        unsigned long long g = (unsigned long long)(uintptr_t)(bStage + (size_t)kt * 1024);
        unsigned l = smemBase + buf * 2048 + t * 8;
        asm volatile("global_load_async_to_lds_b64 %0, %1, off"
                     :: "v"(l), "v"(g) : "memory");
    };

    v8i acc00 = {}, acc01 = {}, acc10 = {}, acc11 = {};

#define GEMM_STEP(KT)                                                          \
    do {                                                                       \
        const char* bp = smem + ((KT) & 1) * 2048 + (size_t)lane * 32;         \
        v8i a0 = *(const v8i*)(aBase0 + (size_t)(KT) * 1024);                  \
        v8i a1 = *(const v8i*)(aBase1 + (size_t)(KT) * 1024);                  \
        v8i b0 = *(const v8i*)(bp);                                            \
        v8i b1 = *(const v8i*)(bp + 1024);                                     \
        acc00 = __builtin_amdgcn_wmma_i32_16x16x64_iu8(true, a0, true, b0,     \
                                                       acc00, false, false);   \
        acc01 = __builtin_amdgcn_wmma_i32_16x16x64_iu8(true, a0, true, b1,     \
                                                       acc01, false, false);   \
        acc10 = __builtin_amdgcn_wmma_i32_16x16x64_iu8(true, a1, true, b0,     \
                                                       acc10, false, false);   \
        acc11 = __builtin_amdgcn_wmma_i32_16x16x64_iu8(true, a1, true, b1,     \
                                                       acc11, false, false);   \
    } while (0)

    stage(0, 0);
#pragma unroll 4
    for (int kt = 0; kt < 31; ++kt) {
        stage(kt + 1, (kt + 1) & 1);
        __builtin_prefetch(aBase0 + (size_t)(kt + 8) * 1024, 0, 1);
        asm volatile("s_wait_asynccnt 0x1" ::: "memory");
        __syncthreads();
        GEMM_STEP(kt);
        __syncthreads();
    }
    asm volatile("s_wait_asynccnt 0x0" ::: "memory");
    __syncthreads();
    GEMM_STEP(31);
#undef GEMM_STEP

    float s_a  = scale_a_from_bits(*amaxBits);
    int laneHi = lane >> 4;
    int nn     = lane & 15;
#pragma unroll
    for (int i = 0; i < 2; ++i) {
        int mBase = (mt0 + i) * 16;
#pragma unroll
        for (int r = 0; r < 8; ++r) {
            int co   = mBase + r + 8 * laneHi;
            float sc = s_a * s_w[co];
            float bq = rintf(bias[co] / sc) * sc;
#pragma unroll
            for (int jn = 0; jn < 2; ++jn) {
                int n = (nt0 + jn) * 16 + nn;
                int b = n >> 10, p = n & 1023;
                float v = (i == 0) ? (jn == 0 ? (float)acc00[r] : (float)acc01[r])
                                   : (jn == 0 ? (float)acc10[r] : (float)acc11[r]);
                out[(size_t)b * 262144 + (size_t)co * 1024 + p] = v * sc + bq;
            }
        }
    }
}

extern "C" void kernel_launch(void* const* d_in, const int* in_sizes, int n_in,
                              void* d_out, int out_size, void* d_ws, size_t ws_size,
                              hipStream_t stream) {
    const float* x    = (const float*)d_in[0];
    const float* w    = (const float*)d_in[1];
    const float* bias = (const float*)d_in[2];
    float* out = (float*)d_out;
    char*  ws  = (char*)d_ws;

    unsigned* amaxBits = (unsigned*)(ws + WS_AMAX);
    float*    sw       = (float*)(ws + WS_SW);
    char*     Aq       = ws + WS_A;
    char*     Xq       = ws + WS_X;

    init_kernel<<<1, 1, 0, stream>>>(amaxBits);
    amax_kernel<<<2048, 256, 0, stream>>>(x, amaxBits, 16777216 / 4);
    wquant_kernel<<<256, 256, 0, stream>>>(w, sw, Aq);
    xquant_kernel<<<16384, 256, 0, stream>>>(x, amaxBits, Xq);
    gemm_iu8_kernel<<<256, 256, 0, stream>>>(Aq, Xq, sw, bias, amaxBits, out);
}
